// TransformerEnhancedGCN_15848429322686
// MI455X (gfx1250) — matrix operand order repaired and checked by reading, not compiled, not run on hardware
//
#include <hip/hip_runtime.h>
#include <cstdint>
#include <cstddef>

// ---------------------------------------------------------------------------
// Types for CDNA5 WMMA (wave32, 16x16x32 f16 -> f32 accum)
// ---------------------------------------------------------------------------
typedef __attribute__((ext_vector_type(16))) _Float16 v16h;
typedef __attribute__((ext_vector_type(8)))  _Float16 v8h;
typedef __attribute__((ext_vector_type(8)))  float    v8f;

#define EPSF 1e-5f

__device__ __forceinline__ v8f wmma_f16(v16h a, v16h b, v8f c) {
  return __builtin_amdgcn_wmma_f32_16x16x32_f16(false, a, false, b, (short)0, c,
                                                false, false);
}

// ---------------------------------------------------------------------------
// Generic WMMA GEMM:  C[M x N] = A[M x K] (f16, row-major, lda)
//                              @ W[N x K]^T (f16, row-major, ldw)
// Epilogue: y = acc*scale[n] + shift[n]; act: 0=none, 1=relu, 2=gelu(exact)
// Writes f32 (Cf) and/or f16 (Ch), both with leading dim ldc, cols < Nvalid.
// Each wave computes a 16 x (16*NTILES) strip; grid-stride over strips.
// A-fragment: lane r=lane&15 holds row m0+r; K-chunk [k0+grp*16, +16)
// (grp = lane>>4)
// B-fragment: lane r holds W row n0+r, K-chunk [k0+grp*16, +16)
// C-fragment: elem i -> row m0 + i + 8*grp, col n0 + r
// ---------------------------------------------------------------------------
template <int NTILES>
__global__ void __launch_bounds__(256) wmma_gemm_k(
    const _Float16* __restrict__ A, int lda,
    const _Float16* __restrict__ W, int ldw,
    const float* __restrict__ scale, const float* __restrict__ shift,
    float* __restrict__ Cf, _Float16* __restrict__ Ch, int ldc,
    int M, int ntilesN, int Nvalid, int K, int act) {
  const int lane = threadIdx.x & 31;
  const int r    = lane & 15;
  const int grp  = lane >> 4;
  const int wave = blockIdx.x * (blockDim.x >> 5) + (threadIdx.x >> 5);
  const int nw   = gridDim.x * (blockDim.x >> 5);
  const int spr  = ntilesN / NTILES;          // strips per row of tiles
  const int total = (M >> 4) * spr;
  for (int s = wave; s < total; s += nw) {
    const int m0 = (s / spr) << 4;
    const int n0 = (s % spr) * (NTILES * 16);
    v8f acc[NTILES];
#pragma unroll
    for (int t = 0; t < NTILES; ++t)
#pragma unroll
      for (int i = 0; i < 8; ++i) acc[t][i] = 0.f;
    const _Float16* arow = A + (size_t)(m0 + r) * lda;
    for (int k0 = 0; k0 < K; k0 += 32) {
      v8h alo = *(const v8h*)(arow + k0 + grp * 16);
      v8h ahi = *(const v8h*)(arow + k0 + 8 + grp * 16);
      v16h af;
#pragma unroll
      for (int i = 0; i < 8; ++i) { af[i] = alo[i]; af[i + 8] = ahi[i]; }
#pragma unroll
      for (int t = 0; t < NTILES; ++t) {
        v16h bf = *(const v16h*)(W + (size_t)(n0 + t * 16 + r) * ldw + k0 + grp * 16);
        acc[t] = wmma_f16(af, bf, acc[t]);
      }
    }
#pragma unroll
    for (int t = 0; t < NTILES; ++t) {
      const int nn = n0 + t * 16 + r;
      if (nn >= Nvalid) continue;   // store masking only; no wmma below
      const float sc_ = scale ? scale[nn] : 1.f;
      const float sh_ = shift ? shift[nn] : 0.f;
#pragma unroll
      for (int i = 0; i < 8; ++i) {
        const int mm = m0 + i + grp * 8;
        float v = acc[t][i] * sc_ + sh_;
        if (act == 1)      v = fmaxf(v, 0.f);
        else if (act == 2) v = 0.5f * v * (1.f + erff(v * 0.70710678118654752f));
        const size_t off = (size_t)mm * ldc + nn;
        if (Cf) Cf[off] = v;
        if (Ch) Ch[off] = (_Float16)v;
      }
    }
  }
}

// ---------------------------------------------------------------------------
// LayerNorm (row-wise over C columns), optional pre-LN bias add and ReLU.
// One wave per row; C in {64,128}. Writes f32 and/or f16.
// ---------------------------------------------------------------------------
__global__ void __launch_bounds__(256) ln_kernel(
    const float* __restrict__ X, const float* __restrict__ bias,
    const float* __restrict__ g, const float* __restrict__ b,
    float* __restrict__ Yf, _Float16* __restrict__ Yh,
    int M, int C, int relu) {
  const int lane  = threadIdx.x & 31;
  const int w0    = blockIdx.x * (blockDim.x >> 5) + (threadIdx.x >> 5);
  const int strid = gridDim.x * (blockDim.x >> 5);
  const int per   = C >> 5;
  for (int m = w0; m < M; m += strid) {
    const float* x = X + (size_t)m * C;
    float sum = 0.f, sq = 0.f;
    for (int j = 0; j < per; ++j) {
      int c = lane + (j << 5);
      float v = x[c] + (bias ? bias[c] : 0.f);
      sum += v; sq += v * v;
    }
    for (int msk = 16; msk; msk >>= 1) {
      sum += __shfl_xor(sum, msk, 32);
      sq  += __shfl_xor(sq,  msk, 32);
    }
    const float mean = sum / C;
    const float inv  = rsqrtf(sq / C - mean * mean + EPSF);
    for (int j = 0; j < per; ++j) {
      int c = lane + (j << 5);
      float v = x[c] + (bias ? bias[c] : 0.f);
      float y = (v - mean) * inv * g[c] + b[c];
      if (relu) y = fmaxf(y, 0.f);
      if (Yf) Yf[(size_t)m * C + c] = y;
      if (Yh) Yh[(size_t)m * C + c] = (_Float16)y;
    }
  }
}

// ---------------------------------------------------------------------------
// Attention: one block per (batch, head, 32-row chunk). Scores in LDS f32,
// V staged in LDS f16.  S = (q/sqrt(DH)) K^T via WMMA, softmax in LDS,
// O = P V via WMMA.  qkv: f16 [NT x 384] (q|k|v), o: f16 [NT x 128].
// ---------------------------------------------------------------------------
__global__ void __launch_bounds__(256) attn_kernel(
    const _Float16* __restrict__ qkv, _Float16* __restrict__ o) {
  __shared__ float    S[32 * 256];   // 32 KB
  __shared__ _Float16 Vs[256 * 32];  // 16 KB
  const int tid   = threadIdx.x;
  const int lane  = tid & 31;
  const int wave  = tid >> 5;
  const int r     = lane & 15;
  const int grp   = lane >> 4;
  const int chunk = blockIdx.x & 7;        // 8 chunks of 32 rows
  const int bh    = blockIdx.x >> 3;
  const int head  = bh & 3;
  const int b     = bh >> 2;
  const _Float16* base = qkv + (size_t)b * 256 * 384;
  const _Float16* qb = base + head * 32;         // already scaled by 1/sqrt(DH)
  const _Float16* kb = base + 128 + head * 32;
  const _Float16* vb = base + 256 + head * 32;
  // stage V: 256 threads x 1 row (32 halves)
  {
    const _Float16* vr = vb + (size_t)tid * 384;
    _Float16* dr = Vs + tid * 32;
#pragma unroll
    for (int j = 0; j < 32; j += 8) *(v8h*)(dr + j) = *(const v8h*)(vr + j);
  }
  // phase 1: S tiles (2 row-tiles x 16 col-tiles = 32 tiles, 4 per wave)
  for (int tt = wave * 4; tt < wave * 4 + 4; ++tt) {
    const int lr0 = (tt >> 4) * 16;        // 0 or 16 (local row-tile)
    const int m0  = chunk * 32 + lr0;      // global query row base
    const int n0  = (tt & 15) * 16;        // key column base
    const _Float16* arow = qb + (size_t)(m0 + r) * 384;
    v8h alo = *(const v8h*)(arow + grp * 16);
    v8h ahi = *(const v8h*)(arow + 8 + grp * 16);
    v16h af;
#pragma unroll
    for (int i = 0; i < 8; ++i) { af[i] = alo[i]; af[i + 8] = ahi[i]; }
    v16h bf = *(const v16h*)(kb + (size_t)(n0 + r) * 384 + grp * 16);
    v8f acc;
#pragma unroll
    for (int i = 0; i < 8; ++i) acc[i] = 0.f;
    acc = wmma_f16(af, bf, acc);
#pragma unroll
    for (int i = 0; i < 8; ++i)
      S[(lr0 + i + grp * 8) * 256 + n0 + r] = acc[i];
  }
  __syncthreads();
  // phase 2: softmax, one thread per row (rows 0..31)
  if (tid < 32) {
    float* row = S + tid * 256;
    float mx = -1e30f;
    for (int j = 0; j < 256; ++j) mx = fmaxf(mx, row[j]);
    float sum = 0.f;
    for (int j = 0; j < 256; ++j) { float e = __expf(row[j] - mx); row[j] = e; sum += e; }
    const float inv = 1.f / sum;
    for (int j = 0; j < 256; ++j) row[j] *= inv;
  }
  __syncthreads();
  // phase 3: O = P V  (2 row-tiles x 2 col-tiles = 4 tiles; waves 0..3)
  if (wave < 4) {
    const int rt = wave >> 1, ct = wave & 1;
    v8f acc;
#pragma unroll
    for (int i = 0; i < 8; ++i) acc[i] = 0.f;
    const float* prow = S + (rt * 16 + r) * 256;
    for (int k0 = 0; k0 < 256; k0 += 32) {
      v16h af;
#pragma unroll
      for (int i = 0; i < 8; ++i) {
        af[i]     = (_Float16)prow[k0 + grp * 16 + i];
        af[i + 8] = (_Float16)prow[k0 + 8 + grp * 16 + i];
      }
      v16h bf;
#pragma unroll
      for (int i = 0; i < 16; ++i)
        bf[i] = Vs[(k0 + grp * 16 + i) * 32 + ct * 16 + r];
      acc = wmma_f16(af, bf, acc);
    }
#pragma unroll
    for (int i = 0; i < 8; ++i) {
      const size_t mm = (size_t)b * 256 + chunk * 32 + rt * 16 + i + 8 * grp;
      o[mm * 128 + head * 32 + ct * 16 + r] = (_Float16)acc[i];
    }
  }
}

// ---------------------------------------------------------------------------
// Small elementwise / prep kernels
// ---------------------------------------------------------------------------
__global__ void fill_f32(float* p, float v, size_t n) {
  size_t i = (size_t)blockIdx.x * blockDim.x + threadIdx.x;
  if (i < n) p[i] = v;
}
__global__ void addto_f32(float* d, const float* s, size_t n) {
  size_t i = (size_t)blockIdx.x * blockDim.x + threadIdx.x;
  if (i < n) d[i] += s[i];
}
__global__ void cvt_f16(const float* s, _Float16* d, size_t n) {
  size_t i = (size_t)blockIdx.x * blockDim.x + threadIdx.x;
  if (i < n) d[i] = (_Float16)s[i];
}
// f32 [rs x cs] -> f16 [rd x cd], zero padded
__global__ void convert_pad(const float* __restrict__ src, _Float16* __restrict__ dst,
                            int rs, int cs, int rd, int cd) {
  int idx = blockIdx.x * blockDim.x + threadIdx.x;
  if (idx >= rd * cd) return;
  int r = idx / cd, c = idx % cd;
  float v = (r < rs && c < cs) ? src[r * cs + c] : 0.f;
  dst[idx] = (_Float16)v;
}
// BN0 on geom (9 cols) -> f16 padded to 32 cols
__global__ void bn0_pad(const float* __restrict__ geom,
                        const float* __restrict__ g, const float* __restrict__ b,
                        const float* __restrict__ rm, const float* __restrict__ rv,
                        _Float16* __restrict__ out, int NT) {
  int idx = blockIdx.x * blockDim.x + threadIdx.x;
  if (idx >= NT * 32) return;
  int r = idx >> 5, c = idx & 31;
  float v = 0.f;
  if (c < 9) {
    float x = geom[r * 9 + c];
    v = (x - rm[c]) * rsqrtf(rv[c] + EPSF) * g[c] + b[c];
  }
  out[idx] = (_Float16)v;
}
// Per-column epilogue affine. mode 0: y=acc+lb  |  mode 1: BN-fold after bias
// mode 2: qkv (scale q section incl. bias by 1/sqrt(DH))
__global__ void prep_affine(float* sc, float* sh, const float* lb, const float* g,
                            const float* bt, const float* rm, const float* rv,
                            int n, int mode) {
  int i = blockIdx.x * blockDim.x + threadIdx.x;
  if (i >= n) return;
  float lbi = lb ? lb[i] : 0.f;
  if (mode == 0) { sc[i] = 1.f; sh[i] = lbi; }
  else if (mode == 1) {
    float s = g[i] * rsqrtf(rv[i] + EPSF);
    sc[i] = s; sh[i] = (lbi - rm[i]) * s + bt[i];
  } else {
    float q = (i < 128) ? 0.17677669529663687f : 1.f;  // 1/sqrt(32)
    sc[i] = q; sh[i] = lbi * q;
  }
}
// ---------------------------------------------------------------------------
// GCN support
// ---------------------------------------------------------------------------
__global__ void deg_count(const int* __restrict__ dst, float* __restrict__ deg, int E0) {
  int e = blockIdx.x * blockDim.x + threadIdx.x;
  if (e < E0) atomicAdd(&deg[dst[e]], 1.0f);
}
__global__ void deg_inv(float* deg, int NT) {
  int i = blockIdx.x * blockDim.x + threadIdx.x;
  if (i < NT) deg[i] = rsqrtf(deg[i] + 1.0f);   // +1 self loop
}
// agg[dst] += hw[src] * dinv[src]*dinv[dst]   (4 cols per thread)
__global__ void gcn_scatter(const float* __restrict__ hw, const int* __restrict__ src,
                            const int* __restrict__ dst, const float* __restrict__ dinv,
                            float* __restrict__ agg, int E0) {
  int idx = blockIdx.x * blockDim.x + threadIdx.x;
  int e = idx >> 5;
  if (e >= E0) return;
  int c = (idx & 31) * 4;
  int s = src[e], d = dst[e];
  float w = dinv[s] * dinv[d];
  const float* hp = hw + (size_t)s * 128 + c;
  float* ap = agg + (size_t)d * 128 + c;
  atomicAdd(ap + 0, hp[0] * w);
  atomicAdd(ap + 1, hp[1] * w);
  atomicAdd(ap + 2, hp[2] * w);
  atomicAdd(ap + 3, hp[3] * w);
}
__global__ void gcn_self(const float* __restrict__ hw, const float* __restrict__ dinv,
                         float* __restrict__ agg, int NT) {
  int idx = blockIdx.x * blockDim.x + threadIdx.x;
  int i = idx >> 5;
  if (i >= NT) return;
  int c = (idx & 31) * 4;
  float w = dinv[i] * dinv[i];
  size_t o = (size_t)i * 128 + c;
#pragma unroll
  for (int j = 0; j < 4; ++j) agg[o + j] += hw[o + j] * w;
}

// ---------------------------------------------------------------------------
// Host launcher
// ---------------------------------------------------------------------------
static inline unsigned nb(size_t n) { return (unsigned)((n + 255) / 256); }

static void gemm(hipStream_t st, const _Float16* A, int lda, const _Float16* W, int ldw,
                 const float* sc, const float* sh, float* Cf, _Float16* Ch, int ldc,
                 int M, int N, int Nvalid, int K, int act) {
  int tn = N / 16;
  if ((tn & 3) == 0)
    wmma_gemm_k<4><<<1024, 256, 0, st>>>(A, lda, W, ldw, sc, sh, Cf, Ch, ldc, M, tn, Nvalid, K, act);
  else
    wmma_gemm_k<1><<<1024, 256, 0, st>>>(A, lda, W, ldw, sc, sh, Cf, Ch, ldc, M, tn, Nvalid, K, act);
}

extern "C" void kernel_launch(void* const* d_in, const int* in_sizes, int n_in,
                              void* d_out, int out_size, void* d_ws, size_t ws_size,
                              hipStream_t stream) {
  const int NT = 65536, H = 128;
  if (n_in < 68) return;
  auto F = [&](int k) { return (const float*)d_in[k]; };
  const float* geom = F(0);
  const int* ei = (const int*)d_in[1];
  const int E0 = in_sizes[1] / 2;
  const int* esrc = ei;
  const int* edst = ei + E0;

  // ---- input index map (setup_inputs insertion order) ----
  // 2-5 bn0{g,b,rm,rv}; 6 W1; 7 b1; 8-11 bn1; 12 W2; 13 b2; 14-17 bn2;
  // 18 W3; 19 b3; 20-21 lnf{g,b};
  // 22 gW1 23 gb1 24-25 gln1; 26..29; 30..33;
  // 34 + 12*l: ln1g,ln1b,Wqkv,bqkv,Wo,bo,ln2g,ln2b,Wf1,bf1,Wf2,bf2
  // 58 hW1 59 hb1 60-61 hln1; 62 hW2 63 hb2 64-65 hln2; 66 hW3 67 hb3

  // ---- workspace bump allocator ----
  char* ws = (char*)d_ws;
  size_t off = 0;
  auto alloc = [&](size_t bytes) -> void* {
    off = (off + 255) & ~(size_t)255;
    void* p = ws + off;
    off += bytes;
    return p;
  };
  float*     x0   = (float*)alloc((size_t)NT * H * 4);        // residual / d
  float*     t0   = (float*)alloc((size_t)NT * H * 4);        // f32 scratch
  char*      big  = (char*)alloc((size_t)NT * 512 * 2);       // 64MB multi-use
  _Float16*  yh   = (_Float16*)alloc((size_t)NT * H * 2);     // f16 buf F
  _Float16*  hh   = (_Float16*)alloc((size_t)NT * H * 2);     // f16 buf G
  float*     deg  = (float*)alloc((size_t)NT * 4);
  float*     sc   = (float*)alloc(512 * 4);
  float*     sh   = (float*)alloc(512 * 4);
  _Float16*  wbuf = (_Float16*)alloc((size_t)512 * 128 * 2);  // staged f16 weight
  (void)ws_size;

  _Float16* bigh = (_Float16*)big;
  _Float16* xbh  = bigh;                     // NT x 32  (FE input)
  _Float16* h1h  = bigh + (size_t)NT * 32;   // NT x 64  (FE hidden)
  float*    agg  = (float*)big;              // NT x 128 (GCN aggregate)

  // ================= feature extractor =================
  bn0_pad<<<nb((size_t)NT * 32), 256, 0, stream>>>(geom, F(2), F(3), F(4), F(5), xbh, NT);
  // Lin1 (9->64) + BN1 + ReLU
  convert_pad<<<nb(64 * 32), 256, 0, stream>>>(F(6), wbuf, 64, 9, 64, 32);
  prep_affine<<<1, 64, 0, stream>>>(sc, sh, F(7), F(8), F(9), F(10), F(11), 64, 1);
  gemm(stream, xbh, 32, wbuf, 32, sc, sh, nullptr, h1h, 64, NT, 64, 64, 32, 1);
  // Lin2 (64->128) + BN2 + ReLU
  convert_pad<<<nb(128 * 64), 256, 0, stream>>>(F(12), wbuf, 128, 64, 128, 64);
  prep_affine<<<1, 128, 0, stream>>>(sc, sh, F(13), F(14), F(15), F(16), F(17), 128, 1);
  gemm(stream, h1h, 64, wbuf, 64, sc, sh, nullptr, hh, 128, NT, 128, 128, 64, 1);
  // Lin3 (128->128) + bias
  convert_pad<<<nb(128 * 128), 256, 0, stream>>>(F(18), wbuf, 128, 128, 128, 128);
  prep_affine<<<1, 128, 0, stream>>>(sc, sh, F(19), nullptr, nullptr, nullptr, nullptr, 128, 0);
  gemm(stream, hh, 128, wbuf, 128, sc, sh, t0, nullptr, 128, NT, 128, 128, 128, 0);
  // LN(lnf): f32 -> x0 (saved), f16 -> yh (GCN input)
  ln_kernel<<<4096, 256, 0, stream>>>(t0, nullptr, F(20), F(21), x0, yh, NT, 128, 0);

  // ================= GCN degree / dinv =================
  fill_f32<<<nb(NT), 256, 0, stream>>>(deg, 0.f, NT);
  deg_count<<<nb(E0), 256, 0, stream>>>(edst, deg, E0);
  deg_inv<<<nb(NT), 256, 0, stream>>>(deg, NT);   // deg now = dinv

  // ================= 3 GCN layers =================
  const int gWi[3] = {22, 26, 30};
  for (int i = 0; i < 3; ++i) {
    convert_pad<<<nb(128 * 128), 256, 0, stream>>>(F(gWi[i]), wbuf, 128, 128, 128, 128);
    gemm(stream, yh, 128, wbuf, 128, nullptr, nullptr, t0, nullptr, 128, NT, 128, 128, 128, 0);
    fill_f32<<<nb((size_t)NT * 128), 256, 0, stream>>>(agg, 0.f, (size_t)NT * 128);
    gcn_scatter<<<nb((size_t)E0 * 32), 256, 0, stream>>>(t0, esrc, edst, deg, agg, E0);
    gcn_self<<<nb((size_t)NT * 32), 256, 0, stream>>>(t0, deg, agg, NT);
    // relu(LN(agg + gb)); f16 -> yh (next layer), f32 -> t0 (residual at i==2)
    ln_kernel<<<4096, 256, 0, stream>>>(agg, F(gWi[i] + 1), F(gWi[i] + 2), F(gWi[i] + 3),
                                        t0, yh, NT, 128, 1);
  }
  // residual: d = x0 + gcn_out   (in place in x0)
  addto_f32<<<nb((size_t)NT * 128), 256, 0, stream>>>(x0, t0, (size_t)NT * 128);

  // ================= 2 transformer layers =================
  for (int l = 0; l < 2; ++l) {
    const int L = 34 + 12 * l;
    // y = LN1(d) -> f16
    ln_kernel<<<4096, 256, 0, stream>>>(x0, nullptr, F(L), F(L + 1), nullptr, yh, NT, 128, 0);
    // qkv = y @ Wqkv^T + bqkv, q-section pre-scaled by 1/sqrt(DH)
    convert_pad<<<nb(384 * 128), 256, 0, stream>>>(F(L + 2), wbuf, 384, 128, 384, 128);
    prep_affine<<<2, 256, 0, stream>>>(sc, sh, F(L + 3), nullptr, nullptr, nullptr, nullptr, 384, 2);
    gemm(stream, yh, 128, wbuf, 128, sc, sh, nullptr, bigh, 384, NT, 384, 384, 128, 0);
    // attention -> hh (f16, NT x 128)
    attn_kernel<<<256 * 4 * 8, 256, 0, stream>>>(bigh, hh);
    // o @ Wo^T + bo -> t0; d += t0
    convert_pad<<<nb(128 * 128), 256, 0, stream>>>(F(L + 4), wbuf, 128, 128, 128, 128);
    prep_affine<<<1, 128, 0, stream>>>(sc, sh, F(L + 5), nullptr, nullptr, nullptr, nullptr, 128, 0);
    gemm(stream, hh, 128, wbuf, 128, sc, sh, t0, nullptr, 128, NT, 128, 128, 128, 0);
    addto_f32<<<nb((size_t)NT * 128), 256, 0, stream>>>(x0, t0, (size_t)NT * 128);
    // y = LN2(d) -> f16
    ln_kernel<<<4096, 256, 0, stream>>>(x0, nullptr, F(L + 6), F(L + 7), nullptr, yh, NT, 128, 0);
    // FFN: gelu(y @ Wf1^T + bf1) -> f16 hidden (NT x 512)
    convert_pad<<<nb(512 * 128), 256, 0, stream>>>(F(L + 8), wbuf, 512, 128, 512, 128);
    prep_affine<<<2, 256, 0, stream>>>(sc, sh, F(L + 9), nullptr, nullptr, nullptr, nullptr, 512, 0);
    gemm(stream, yh, 128, wbuf, 128, sc, sh, nullptr, bigh, 512, NT, 512, 512, 128, 2);
    // hidden @ Wf2^T + bf2 -> t0; d += t0
    convert_pad<<<nb(128 * 512), 256, 0, stream>>>(F(L + 10), wbuf, 128, 512, 128, 512);
    prep_affine<<<1, 128, 0, stream>>>(sc, sh, F(L + 11), nullptr, nullptr, nullptr, nullptr, 128, 0);
    gemm(stream, bigh, 512, wbuf, 512, sc, sh, t0, nullptr, 128, NT, 128, 128, 512, 0);
    addto_f32<<<nb((size_t)NT * 128), 256, 0, stream>>>(x0, t0, (size_t)NT * 128);
  }

  // ================= MLP head =================
  cvt_f16<<<nb((size_t)NT * 128), 256, 0, stream>>>(x0, yh, (size_t)NT * 128);
  // h1: (128->128) + bias -> LN(hln1) + ReLU
  convert_pad<<<nb(128 * 128), 256, 0, stream>>>(F(58), wbuf, 128, 128, 128, 128);
  prep_affine<<<1, 128, 0, stream>>>(sc, sh, F(59), nullptr, nullptr, nullptr, nullptr, 128, 0);
  gemm(stream, yh, 128, wbuf, 128, sc, sh, t0, nullptr, 128, NT, 128, 128, 128, 0);
  ln_kernel<<<4096, 256, 0, stream>>>(t0, nullptr, F(60), F(61), nullptr, yh, NT, 128, 1);
  // h2: (128->64) + bias -> LN(hln2) + ReLU
  convert_pad<<<nb(64 * 128), 256, 0, stream>>>(F(62), wbuf, 64, 128, 64, 128);
  prep_affine<<<1, 64, 0, stream>>>(sc, sh, F(63), nullptr, nullptr, nullptr, nullptr, 64, 0);
  gemm(stream, yh, 128, wbuf, 128, sc, sh, t0, nullptr, 64, NT, 64, 64, 128, 0);
  ln_kernel<<<4096, 256, 0, stream>>>(t0, nullptr, F(64), F(65), nullptr, yh, NT, 64, 1);
  // h3: (64->10), N padded to 16, write d_out (f32, NT x 10)
  convert_pad<<<nb(16 * 64), 256, 0, stream>>>(F(66), wbuf, 10, 64, 16, 64);
  prep_affine<<<1, 32, 0, stream>>>(sc, sh, F(67), nullptr, nullptr, nullptr, nullptr, 10, 0);
  gemm(stream, yh, 64, wbuf, 64, sc, sh, (float*)d_out, nullptr, 10, NT, 16, 10, 64, 0);
  (void)out_size;
}
